// HeteroGNN_3075196584237
// MI455X (gfx1250) — matrix operand order repaired and checked by reading, not compiled, not run on hardware
//
#include <hip/hip_runtime.h>
#include <hip/hip_bf16.h>

typedef __attribute__((ext_vector_type(16))) __bf16 v16bf;
typedef __attribute__((ext_vector_type(8)))  __bf16 bf16x8;
typedef __attribute__((ext_vector_type(4)))  __bf16 bf16x4;
typedef __attribute__((ext_vector_type(8)))  float  v8f;
typedef __attribute__((ext_vector_type(4)))  float  f32x4;

union VFrag { v16bf v; bf16x8 h[2]; };

#define NP   100000
#define ND   2000
#define NE   1000
#define EDG  400000
#define DH   64
#define NEG_SLOPE 0.2f

// ---------------------------------------------------------------------------
// utility kernels
// ---------------------------------------------------------------------------
__global__ void k_clear_u32(unsigned* __restrict__ p, unsigned v, int n) {
  int i = blockIdx.x * blockDim.x + threadIdx.x;
  if (i < n) p[i] = v;
}

// c[k] = sum_j W[k,j] * a[j]   (W is [din,64]); folds Ws@a_s so al = X @ c
__global__ void k_cvec(const float* __restrict__ W, const float* __restrict__ a,
                       float* __restrict__ c, int din) {
  int k = threadIdx.x;
  if (k < din) {
    float s = 0.f;
    for (int j = 0; j < DH; ++j) s += W[k * DH + j] * a[j];
    c[k] = s;
  }
}

// one wave per row: al[i] = dot(X[i,0:din], c)
__global__ void k_rowdot(const float* __restrict__ X, const float* __restrict__ c,
                         float* __restrict__ al, int N, int din) {
  int wid  = (blockIdx.x * blockDim.x + threadIdx.x) >> 5;
  int lane = threadIdx.x & 31;
  if (wid >= N) return;
  float s = 0.f;
  for (int k = lane; k < din; k += 32) s += X[(size_t)wid * din + k] * c[k];
  for (int off = 16; off; off >>= 1) s += __shfl_down(s, off, 32);
  if (lane == 0) al[wid] = s;
}

// ---------------------------------------------------------------------------
// WMMA GEMM: H[Ns x 64] = X[Ns x din] @ W[din x 64]  (bf16 in, f32 acc)
// block = 256 threads = 8 waves; wave -> 16 rows; block -> 128 rows.
// K padded to 64 in LDS (din in {16,32,64}).
// Fragments fetched as contiguous 16B chunks -> ds_load_b128.
// ---------------------------------------------------------------------------
__global__ void __launch_bounds__(256)
k_gemm_hs(const float* __restrict__ X, const float* __restrict__ W,
          float* __restrict__ H, int Ns, int din) {
  __shared__ alignas(16) __bf16 sX[128 * 64];   // [row][K]   row-major
  __shared__ alignas(16) __bf16 sWt[64 * 64];   // [N][K]     transposed
  const int tid  = threadIdx.x;
  const int row0 = blockIdx.x * 128;
  for (int idx = tid; idx < 64 * 64; idx += 256) {          // W^T: [n][k]
    int n = idx >> 6, k = idx & 63;
    sWt[idx] = (k < din) ? (__bf16)W[k * DH + n] : (__bf16)0.f;
  }
  if (din == 64) {                                          // vectorized X staging
    for (int chunk = tid; chunk < 128 * 16; chunk += 256) {
      int r = chunk >> 4, c4 = chunk & 15;
      int gr = row0 + r;
      f32x4 v = {};
      if (gr < Ns) v = *(const f32x4*)(X + (size_t)gr * 64 + c4 * 4);
      bf16x4 h = {(__bf16)v.x, (__bf16)v.y, (__bf16)v.z, (__bf16)v.w};
      *(bf16x4*)(&sX[r * 64 + c4 * 4]) = h;
    }
  } else {
    for (int idx = tid; idx < 128 * 64; idx += 256) {       // X: [128 rows][K pad 64]
      int r = idx >> 6, cc = idx & 63;
      int gr = row0 + r;
      sX[idx] = (gr < Ns && cc < din) ? (__bf16)X[(size_t)gr * din + cc] : (__bf16)0.f;
    }
  }
  __syncthreads();

  const int wave = tid >> 5, lane = tid & 31;
  const int m = lane & 15, half = lane >> 4;
  const __bf16* xrow = &sX[(wave * 16 + m) * 64];
  v8f acc[4] = {};
#pragma unroll
  for (int kk = 0; kk < 2; ++kk) {
    VFrag a;                                               // A: two contiguous runs of 8
    a.h[0] = *(const bf16x8*)(xrow + kk * 32 + half * 8);
    a.h[1] = *(const bf16x8*)(xrow + kk * 32 + 16 + half * 8);
#pragma unroll
    for (int t = 0; t < 4; ++t) {
      VFrag b;                                             // B: two contiguous runs of 8
      const __bf16* wcol = &sWt[(t * 16 + m) * 64 + kk * 32];
      b.h[0] = *(const bf16x8*)(wcol + half * 8);
      b.h[1] = *(const bf16x8*)(wcol + 16 + half * 8);
      acc[t] = __builtin_amdgcn_wmma_f32_16x16x32_bf16(
          false, a.v, false, b.v, (short)0, acc[t], false, false);
    }
  }
  const int rbase = row0 + wave * 16 + half * 8;            // f32 C/D 16x16 layout
  if (rbase + 8 <= Ns) {                                    // fast path: no guards
#pragma unroll
    for (int t = 0; t < 4; ++t)
#pragma unroll
      for (int r = 0; r < 8; ++r)
        H[(size_t)(rbase + r) * DH + t * 16 + m] = acc[t][r];
  } else {
#pragma unroll
    for (int t = 0; t < 4; ++t)
#pragma unroll
      for (int r = 0; r < 8; ++r)
        if (rbase + r < Ns) H[(size_t)(rbase + r) * DH + t * 16 + m] = acc[t][r];
  }
}

// ---------------------------------------------------------------------------
// edge-softmax kernels (order-preserving float<->uint encoding for atomic max)
// ---------------------------------------------------------------------------
__device__ inline unsigned enc_f32(float x) {
  unsigned b = __float_as_uint(x);
  return (b & 0x80000000u) ? ~b : (b | 0x80000000u);
}
__device__ inline float dec_f32(unsigned e) {
  unsigned b = (e & 0x80000000u) ? (e ^ 0x80000000u) : ~e;
  return __uint_as_float(b);
}

__global__ void k_edge_logit(const float* __restrict__ al_s, const float* __restrict__ al_d,
                             const int* __restrict__ src, const int* __restrict__ dst,
                             float* __restrict__ ebuf, unsigned* __restrict__ menc, int E) {
  int i = blockIdx.x * blockDim.x + threadIdx.x;
  if (i >= E) return;
  float v = al_s[src[i]] + al_d[dst[i]];
  v = (v > 0.f) ? v : NEG_SLOPE * v;
  ebuf[i] = v;
  atomicMax(&menc[dst[i]], enc_f32(v));
}

__global__ void k_final_m(const unsigned* __restrict__ menc, float* __restrict__ mf, int Nd) {
  int i = blockIdx.x * blockDim.x + threadIdx.x;
  if (i >= Nd) return;
  float mv = dec_f32(menc[i]);      // enc==0 (empty segment) decodes to NaN
  mf[i] = (mv == mv && mv > -3.0e38f && mv < 3.0e38f) ? mv : 0.f;
}

__global__ void k_edge_w(const float* __restrict__ ebuf, const float* __restrict__ mf,
                         const int* __restrict__ dst, float* __restrict__ wbuf,
                         float* __restrict__ denom, int E) {
  int i = blockIdx.x * blockDim.x + threadIdx.x;
  if (i >= E) return;
  float w = __expf(ebuf[i] - mf[dst[i]]);
  wbuf[i] = w;
  atomicAdd(&denom[dst[i]], w);
}

// one wave per edge: out[dst] += (w/denom) * hs[src]  (64 floats, 2 per lane)
__global__ void k_scatter(const float* __restrict__ wbuf, const float* __restrict__ denom,
                          const int* __restrict__ src, const int* __restrict__ dst,
                          const float* __restrict__ hs, float* __restrict__ out, int E) {
  int wid  = (blockIdx.x * blockDim.x + threadIdx.x) >> 5;
  int lane = threadIdx.x & 31;
  if (wid >= E) return;
  int s = src[wid], d = dst[wid];
  float coef = wbuf[wid] / fmaxf(denom[d], 1e-16f);
  const float* hrow = hs + (size_t)s * DH;
  float* orow = out + (size_t)d * DH;
  atomicAdd(&orow[lane],      coef * hrow[lane]);
  atomicAdd(&orow[lane + 32], coef * hrow[lane + 32]);
}

// ---------------------------------------------------------------------------
// combine kernels (HeteroConv mean over relations + bias + ReLU)
// ---------------------------------------------------------------------------
__global__ void k_combine1(const float* __restrict__ o, const float* __restrict__ b,
                           float* __restrict__ xn, int N) {
  int i = blockIdx.x * blockDim.x + threadIdx.x;
  if (i >= N * DH) return;
  float v = o[i] + b[i & 63];
  xn[i] = v > 0.f ? v : 0.f;
}
__global__ void k_combine_pt(const float* __restrict__ o, const float* __restrict__ b1,
                             const float* __restrict__ b2, float* __restrict__ xn, int N) {
  int i = blockIdx.x * blockDim.x + threadIdx.x;
  if (i >= N * DH) return;
  float v = (o[i] + b1[i & 63] + b2[i & 63]) * 0.5f;
  xn[i] = v > 0.f ? v : 0.f;
}

// ---------------------------------------------------------------------------
// final linear: out[Ns x 16] = X[Ns x 64] @ Wl[64 x 16] + bl   (WMMA, 1 N-tile)
// ---------------------------------------------------------------------------
__global__ void __launch_bounds__(256)
k_final_linear(const float* __restrict__ X, const float* __restrict__ Wl,
               const float* __restrict__ bl, float* __restrict__ out, int Ns) {
  __shared__ alignas(16) __bf16 sX[128 * 64];
  __shared__ alignas(16) __bf16 sWt[16 * 64];   // [n][k] transposed
  const int tid = threadIdx.x;
  const int row0 = blockIdx.x * 128;
  for (int idx = tid; idx < 16 * 64; idx += 256) {
    int n = idx >> 6, k = idx & 63;
    sWt[idx] = (__bf16)Wl[k * 16 + n];
  }
  for (int chunk = tid; chunk < 128 * 16; chunk += 256) {   // vectorized X staging
    int r = chunk >> 4, c4 = chunk & 15;
    int gr = row0 + r;
    f32x4 v = {};
    if (gr < Ns) v = *(const f32x4*)(X + (size_t)gr * DH + c4 * 4);
    bf16x4 h = {(__bf16)v.x, (__bf16)v.y, (__bf16)v.z, (__bf16)v.w};
    *(bf16x4*)(&sX[r * 64 + c4 * 4]) = h;
  }
  __syncthreads();
  const int wave = tid >> 5, lane = tid & 31;
  const int m = lane & 15, half = lane >> 4;
  const __bf16* xrow = &sX[(wave * 16 + m) * 64];
  v8f acc = {};
#pragma unroll
  for (int kk = 0; kk < 2; ++kk) {
    VFrag a, b;
    a.h[0] = *(const bf16x8*)(xrow + kk * 32 + half * 8);
    a.h[1] = *(const bf16x8*)(xrow + kk * 32 + 16 + half * 8);
    const __bf16* wcol = &sWt[m * 64 + kk * 32];
    b.h[0] = *(const bf16x8*)(wcol + half * 8);
    b.h[1] = *(const bf16x8*)(wcol + 16 + half * 8);
    acc = __builtin_amdgcn_wmma_f32_16x16x32_bf16(
        false, a.v, false, b.v, (short)0, acc, false, false);
  }
  const int rbase = row0 + wave * 16 + half * 8;
  float bias = bl[m];
  if (rbase + 8 <= Ns) {
#pragma unroll
    for (int r = 0; r < 8; ++r)
      out[(size_t)(rbase + r) * 16 + m] = acc[r] + bias;
  } else {
#pragma unroll
    for (int r = 0; r < 8; ++r)
      if (rbase + r < Ns) out[(size_t)(rbase + r) * 16 + m] = acc[r] + bias;
  }
}

// ---------------------------------------------------------------------------
// host orchestration
// ---------------------------------------------------------------------------
struct RelP { const float *Ws, *Wd, *as_, *ad_, *b; };

// Assumes pytree flattened in source (insertion) order:
//  per relation {Ws, Wd, a_s, a_d, b}; relations {pt_d, d_pt, pt_e, e_pt}; layers {l0, l1}.
static inline RelP relp(void* const* d_in, int layer, int rel) {
  int base = 7 + layer * 20 + rel * 5;
  return RelP{(const float*)d_in[base + 0], (const float*)d_in[base + 1],
              (const float*)d_in[base + 2], (const float*)d_in[base + 3],
              (const float*)d_in[base + 4]};
}

extern "C" void kernel_launch(void* const* d_in, const int* in_sizes, int n_in,
                              void* d_out, int out_size, void* d_ws, size_t ws_size,
                              hipStream_t stream) {
  (void)in_sizes; (void)n_in; (void)out_size; (void)ws_size;

  const float* xP  = (const float*)d_in[0];
  const float* xD  = (const float*)d_in[1];
  const float* xE  = (const float*)d_in[2];
  const int* tsrc  = (const int*)d_in[3];
  const int* tdst  = (const int*)d_in[4];
  const int* esrc  = (const int*)d_in[5];
  const int* edst  = (const int*)d_in[6];
  const float* linW = (const float*)d_in[47];
  const float* linb = (const float*)d_in[48];

  float* ws = (float*)d_ws;
  size_t off = 0;
  auto alloc = [&](size_t n) { float* p = ws + off; off += n; return p; };
  float* xpA   = alloc((size_t)NP * DH);
  float* xdA   = alloc((size_t)ND * DH);
  float* xeA   = alloc((size_t)NE * DH);
  float* xpB   = alloc((size_t)NP * DH);
  float* xdB   = alloc((size_t)ND * DH);
  float* xeB   = alloc((size_t)NE * DH);
  float* outPt = alloc((size_t)NP * DH);
  float* outD  = alloc((size_t)ND * DH);
  float* outE  = alloc((size_t)NE * DH);
  float* hs    = alloc((size_t)NP * DH);
  float* alS   = alloc(NP);
  float* alD   = alloc(NP);
  unsigned* menc = (unsigned*)alloc(NP);
  float* mf    = alloc(NP);
  float* den   = alloc(NP);
  float* ebuf  = alloc(EDG);
  float* wbuf  = alloc(EDG);
  float* cs    = alloc(64);
  float* cd    = alloc(64);

  auto cdiv = [](int a, int b) { return (a + b - 1) / b; };

  auto run_rel = [&](const float* Xs, int Ns, int dins,
                     const float* Xd, int Nd, int dind,
                     const int* src, const int* dst, const RelP& P, float* out) {
    k_clear_u32<<<cdiv(Nd, 256), 256, 0, stream>>>(menc, 0u, Nd);
    k_clear_u32<<<cdiv(Nd, 256), 256, 0, stream>>>((unsigned*)den, 0u, Nd);
    k_cvec<<<1, 64, 0, stream>>>(P.Ws, P.as_, cs, dins);
    k_cvec<<<1, 64, 0, stream>>>(P.Wd, P.ad_, cd, dind);
    k_rowdot<<<cdiv(Ns * 32, 256), 256, 0, stream>>>(Xs, cs, alS, Ns, dins);
    k_rowdot<<<cdiv(Nd * 32, 256), 256, 0, stream>>>(Xd, cd, alD, Nd, dind);
    k_gemm_hs<<<cdiv(Ns, 128), 256, 0, stream>>>(Xs, P.Ws, hs, Ns, dins);
    k_edge_logit<<<cdiv(EDG, 256), 256, 0, stream>>>(alS, alD, src, dst, ebuf, menc, EDG);
    k_final_m<<<cdiv(Nd, 256), 256, 0, stream>>>(menc, mf, Nd);
    k_edge_w<<<cdiv(EDG, 256), 256, 0, stream>>>(ebuf, mf, dst, wbuf, den, EDG);
    k_scatter<<<cdiv(EDG * 32, 256), 256, 0, stream>>>(wbuf, den, src, dst, hs, out, EDG);
  };

  auto run_layer = [&](const float* xp, int dp, const float* xd, int dd,
                       const float* xe, int de, int layer,
                       float* xpn, float* xdn, float* xen) {
    k_clear_u32<<<cdiv(NP * DH, 256), 256, 0, stream>>>((unsigned*)outPt, 0u, NP * DH);
    k_clear_u32<<<cdiv(ND * DH, 256), 256, 0, stream>>>((unsigned*)outD, 0u, ND * DH);
    k_clear_u32<<<cdiv(NE * DH, 256), 256, 0, stream>>>((unsigned*)outE, 0u, NE * DH);
    RelP p0 = relp(d_in, layer, 0);   // pt_d : patient -> drug
    RelP p1 = relp(d_in, layer, 1);   // d_pt : drug -> patient
    RelP p2 = relp(d_in, layer, 2);   // pt_e : patient -> effect
    RelP p3 = relp(d_in, layer, 3);   // e_pt : effect -> patient
    run_rel(xp, NP, dp, xd, ND, dd, tsrc, tdst, p0, outD);
    run_rel(xd, ND, dd, xp, NP, dp, tdst, tsrc, p1, outPt);  // both patient-dst rels
    run_rel(xp, NP, dp, xe, NE, de, esrc, edst, p2, outE);
    run_rel(xe, NE, de, xp, NP, dp, edst, esrc, p3, outPt);  // accumulate into outPt
    k_combine_pt<<<cdiv(NP * DH, 256), 256, 0, stream>>>(outPt, p1.b, p3.b, xpn, NP);
    k_combine1<<<cdiv(ND * DH, 256), 256, 0, stream>>>(outD, p0.b, xdn, ND);
    k_combine1<<<cdiv(NE * DH, 256), 256, 0, stream>>>(outE, p2.b, xen, NE);
  };

  run_layer(xP, 64, xD, 32, xE, 16, 0, xpA, xdA, xeA);
  run_layer(xpA, 64, xdA, 64, xeA, 64, 1, xpB, xdB, xeB);
  k_final_linear<<<cdiv(NP, 128), 256, 0, stream>>>(xpB, linW, linb, (float*)d_out, NP);
}
